// MultiTaskGNN_32212254720646
// MI455X (gfx1250) — compile-verified, matrix operand order checked by reading, not run on hardware
//
#include <hip/hip_runtime.h>

typedef __attribute__((ext_vector_type(16))) _Float16 v16h;
typedef __attribute__((ext_vector_type(8)))  float    v8f;

#define N_NODES   50000
#define N_EDGES   400000
#define E_TOT     (N_EDGES + N_NODES)
#define IN_DIM    128
#define HIDDEN    64
#define HEADS     4
#define HC        256      /* HEADS*HIDDEN */
#define LATENT    32
#define NUM_CLASSES 10
#define NUM_GRAPHS  64
#define NEG_SLOPE 0.2f

// ---------------- helpers: monotonic float<->uint encoding for atomic max ----
__device__ __forceinline__ unsigned encF(float f) {
    unsigned u = __float_as_uint(f);
    return (u & 0x80000000u) ? ~u : (u | 0x80000000u);
}
__device__ __forceinline__ float decF(unsigned e) {
    unsigned u = (e & 0x80000000u) ? (e & 0x7FFFFFFFu) : ~e;
    return __uint_as_float(u);
}

// ---------------- fill ------------------------------------------------------
__global__ void fill_u32(unsigned* __restrict__ p, unsigned v, long n) {
    long i = (long)blockIdx.x * blockDim.x + threadIdx.x;
    if (i < n) p[i] = v;
}

// ---------------- edge_attr mean -------------------------------------------
__global__ void ea_mean_k(const float* __restrict__ ea, float* __restrict__ mean, int E) {
    __shared__ float red[3][256];
    float s0 = 0.f, s1 = 0.f, s2 = 0.f;
    for (int e = threadIdx.x; e < E; e += blockDim.x) {
        s0 += ea[3 * e];
        s1 += ea[3 * e + 1];
        s2 += ea[3 * e + 2];
    }
    red[0][threadIdx.x] = s0; red[1][threadIdx.x] = s1; red[2][threadIdx.x] = s2;
    __syncthreads();
    for (int st = 128; st > 0; st >>= 1) {
        if ((int)threadIdx.x < st) {
            red[0][threadIdx.x] += red[0][threadIdx.x + st];
            red[1][threadIdx.x] += red[1][threadIdx.x + st];
            red[2][threadIdx.x] += red[2][threadIdx.x + st];
        }
        __syncthreads();
    }
    if (threadIdx.x == 0) {
        float inv = 1.0f / (float)E;
        mean[0] = red[0][0] * inv; mean[1] = red[1][0] * inv; mean[2] = red[2][0] * inv;
    }
}

// ---------------- B swizzle: fp32 [K,N] row-major -> CDNA5 f16 B fragments --
// fragment layout: [ks][nt][lane][16 halves], lane holds col = nt*16+(lane&15),
// element i -> K = ks*32 + (lane>>4)*8 + 16*(i>>3) + (i&7)
__global__ void swizzleB(const float* __restrict__ B, _Float16* __restrict__ Bs,
                         int K, int N) {
    int idx = blockIdx.x * blockDim.x + threadIdx.x;
    int total = K * N;                     // one half per (k,n)
    if (idx >= total) return;
    int i    = idx & 15;
    int lane = (idx >> 4) & 31;
    int t    = idx >> 9;                   // ks*tilesN + nt
    int tilesN = N >> 4;
    int ks = t / tilesN, nt = t - ks * tilesN;
    int col = nt * 16 + (lane & 15);
    int k   = ks * 32 + ((lane >> 4) * 8) + 16 * (i >> 3) + (i & 7);
    Bs[idx] = (_Float16)B[(long)k * N + col];
}

// ---------------- WMMA GEMM: C[M,N] = A[M,K](f32) * Bswz(f16) (+bias) -------
// wave computes a 16x(16*NB) strip: one A fragment feeds NB independent
// back-to-back WMMAs (A reuse + no acc RAW hazard between them)
template <int NB>
__global__ void gemm_wmma(const float* __restrict__ A, const _Float16* __restrict__ Bs,
                          const float* __restrict__ bias, float* __restrict__ C,
                          int M, int K, int N) {
    const int lane = threadIdx.x & 31;
    const int wave = threadIdx.x >> 5;
    const int tilesN  = N >> 4;
    const int groupsN = tilesN / NB;
    const int tilesM  = (M + 15) >> 4;
    int g = blockIdx.x * (blockDim.x >> 5) + wave;
    if (g >= tilesM * groupsN) return;               // wave-uniform exit
    const int tm = (g / groupsN) * 16;
    const int gn = (g % groupsN) * NB;               // first N-tile of this strip

    const int row  = tm + (lane & 15);
    const int koff = (lane >> 4) * 8;
    const bool rowOK = row < M;
    const float* arow = A + (long)(rowOK ? row : 0) * K;

    v8f acc[NB];
#pragma unroll
    for (int t = 0; t < NB; ++t) acc[t] = (v8f){};

    const int ksteps = K >> 5;
    for (int ks = 0; ks < ksteps; ++ks) {
        v16h a = {};
        if (rowOK) {
            const int k0 = ks * 32 + koff;
#pragma unroll
            for (int j = 0; j < 8; ++j) {
                a[j]     = (_Float16)arow[k0 + j];
                a[8 + j] = (_Float16)arow[k0 + 16 + j];
            }
        }
#pragma unroll
        for (int t = 0; t < NB; ++t) {
            const v16h b = *(const v16h*)(Bs + ((long)(ks * tilesN + gn + t) * 32 + lane) * 16);
            acc[t] = __builtin_amdgcn_wmma_f32_16x16x32_f16(false, a, false, b,
                                                            (short)0, acc[t], false, false);
        }
    }

#pragma unroll
    for (int t = 0; t < NB; ++t) {
        const int col = (gn + t) * 16 + (lane & 15);
        const float bv = bias ? bias[col] : 0.0f;
#pragma unroll
        for (int r = 0; r < 8; ++r) {
            int orow = tm + r + 8 * (lane >> 4);
            if (orow < M) C[(long)orow * N + col] = acc[t][r] + bv;
        }
    }
}

// ---------------- edge pass 1: logits + segment max -------------------------
// one wave per edge; lane handles 8 consecutive channels (head = lane>>3)
__global__ void edge_logits(const float* __restrict__ xl, const float* __restrict__ xr,
                            const int* __restrict__ src, const int* __restrict__ dst,
                            const float* __restrict__ ea, const float* __restrict__ eam,
                            const float* __restrict__ We, const float* __restrict__ att,
                            float* __restrict__ elog, unsigned* __restrict__ mEnc) {
    __shared__ float sWe[3 * HC];
    __shared__ float sAtt[HC];
    for (int i = threadIdx.x; i < 3 * HC; i += blockDim.x) sWe[i] = We[i];
    for (int i = threadIdx.x; i < HC; i += blockDim.x) sAtt[i] = att[i];
    __syncthreads();

    const int lane = threadIdx.x & 31;
    const int e = blockIdx.x * (blockDim.x >> 5) + (threadIdx.x >> 5);
    if (e >= E_TOT) return;

    int s, d; float a0, a1, a2;
    if (e < N_EDGES) {
        s = src[e]; d = dst[e];
        a0 = ea[3 * e]; a1 = ea[3 * e + 1]; a2 = ea[3 * e + 2];
    } else {
        s = d = e - N_EDGES;
        a0 = eam[0]; a1 = eam[1]; a2 = eam[2];
    }
    const float* xls = xl + (long)s * HC;
    const float* xrd = xr + (long)d * HC;
    const int base = lane * 8;
    float p = 0.f;
#pragma unroll
    for (int j = 0; j < 8; ++j) {
        int c = base + j;
        float v = xls[c] + xrd[c] + a0 * sWe[c] + a1 * sWe[HC + c] + a2 * sWe[2 * HC + c];
        v = v > 0.f ? v : v * NEG_SLOPE;
        p += v * sAtt[c];
    }
    p += __shfl_xor(p, 1, 32);
    p += __shfl_xor(p, 2, 32);
    p += __shfl_xor(p, 4, 32);
    if ((lane & 7) == 0) {
        int h = lane >> 3;
        elog[(long)e * HEADS + h] = p;
        atomicMax(&mEnc[(long)d * HEADS + h], encF(p));
    }
}

// ---------------- edge pass 2: exp + segment denom --------------------------
__global__ void edge_softmax(const int* __restrict__ dst, const unsigned* __restrict__ mEnc,
                             float* __restrict__ elog, float* __restrict__ denom) {
    long idx = (long)blockIdx.x * blockDim.x + threadIdx.x;
    if (idx >= (long)E_TOT * HEADS) return;
    int e = (int)(idx >> 2), h = (int)(idx & 3);
    int d = (e < N_EDGES) ? dst[e] : (e - N_EDGES);
    float m = decF(mEnc[(long)d * HEADS + h]);
    float ex = __expf(elog[idx] - m);
    elog[idx] = ex;
    atomicAdd(&denom[(long)d * HEADS + h], ex);
}

// ---------------- edge pass 3: alpha-weighted scatter -----------------------
__global__ void edge_scatter(const float* __restrict__ xl,
                             const int* __restrict__ src, const int* __restrict__ dst,
                             const float* __restrict__ elog, const float* __restrict__ denom,
                             float* __restrict__ outacc) {
    const int lane = threadIdx.x & 31;
    const int e = blockIdx.x * (blockDim.x >> 5) + (threadIdx.x >> 5);
    if (e >= E_TOT) return;
    int s, d;
    if (e < N_EDGES) { s = src[e]; d = dst[e]; }
    else             { s = d = e - N_EDGES; }
    const int h = lane >> 3;
    float alpha = elog[(long)e * HEADS + h] / denom[(long)d * HEADS + h];
    const float* xls = xl + (long)s * HC + lane * 8;
    float* od = outacc + (long)d * HC + lane * 8;
#pragma unroll
    for (int j = 0; j < 8; ++j) atomicAdd(&od[j], xls[j] * alpha);
}

// ---------------- finalize: head mean + bias + relu -------------------------
__global__ void finalize_h(const float* __restrict__ outacc, const float* __restrict__ b,
                           float* __restrict__ h) {
    long idx = (long)blockIdx.x * blockDim.x + threadIdx.x;
    if (idx >= (long)N_NODES * HIDDEN) return;
    int n = (int)(idx >> 6), c = (int)(idx & 63);
    const float* o = outacc + (long)n * HC;
    float v = 0.25f * (o[c] + o[64 + c] + o[128 + c] + o[192 + c]) + b[c];
    h[idx] = v > 0.f ? v : 0.f;
}

// ---------------- link prediction -------------------------------------------
__global__ void link_k(const float* __restrict__ z, const int* __restrict__ src,
                       const int* __restrict__ dst, const float* __restrict__ w,
                       float* __restrict__ link) {
    int e = blockIdx.x * blockDim.x + threadIdx.x;
    if (e >= N_EDGES) return;
    const float* zs = z + (long)src[e] * LATENT;
    const float* zd = z + (long)dst[e] * LATENT;
    float acc = 0.f;
#pragma unroll
    for (int l = 0; l < LATENT; ++l) acc += zs[l] * zd[l] * w[l];
    link[e] = acc;
}

// ---------------- global add pool -------------------------------------------
__global__ void pool_k(const float* __restrict__ z, const int* __restrict__ batch,
                       float* __restrict__ gemb) {
    long idx = (long)blockIdx.x * blockDim.x + threadIdx.x;
    if (idx >= (long)N_NODES * LATENT) return;
    int n = (int)(idx >> 5), l = (int)(idx & 31);
    atomicAdd(&gemb[batch[n] * LATENT + l], z[idx]);
}

// ---------------- classifier ------------------------------------------------
__global__ void cls_k(const float* __restrict__ gemb, const float* __restrict__ Wc,
                      const float* __restrict__ bc, float* __restrict__ out) {
    int idx = threadIdx.x;
    if (idx >= NUM_GRAPHS * NUM_CLASSES) return;
    int g = idx / NUM_CLASSES, c = idx - g * NUM_CLASSES;
    float acc = bc[c];
#pragma unroll
    for (int l = 0; l < LATENT; ++l) acc += gemb[g * LATENT + l] * Wc[l * NUM_CLASSES + c];
    out[idx] = acc;
}

// ============================================================================
extern "C" void kernel_launch(void* const* d_in, const int* in_sizes, int n_in,
                              void* d_out, int out_size, void* d_ws, size_t ws_size,
                              hipStream_t stream) {
    const float* x     = (const float*)d_in[0];
    const int*   ei    = (const int*)d_in[1];
    const float* eattr = (const float*)d_in[2];
    const int*   batch = (const int*)d_in[3];
    const float* Wl0 = (const float*)d_in[4];
    const float* Wr0 = (const float*)d_in[5];
    const float* We0 = (const float*)d_in[6];
    const float* at0 = (const float*)d_in[7];
    const float* b0  = (const float*)d_in[8];
    const float* Wl1 = (const float*)d_in[9];
    const float* Wr1 = (const float*)d_in[10];
    const float* We1 = (const float*)d_in[11];
    const float* at1 = (const float*)d_in[12];
    const float* b1  = (const float*)d_in[13];
    const float* Wmu = (const float*)d_in[14];
    const float* bmu = (const float*)d_in[15];
    const float* Wde = (const float*)d_in[16];
    const float* bde = (const float*)d_in[17];
    const float* Wcl = (const float*)d_in[18];
    const float* bcl = (const float*)d_in[19];
    const float* wgt = (const float*)d_in[20];

    const int* src = ei;
    const int* dst = ei + N_EDGES;

    // outputs (concatenated flat)
    float* z       = (float*)d_out;
    float* recon   = z + (long)N_NODES * LATENT;
    float* link    = recon + (long)N_NODES * IN_DIM;
    float* logitsO = link + N_EDGES;

    // workspace layout (all 32B aligned)
    char* w = (char*)d_ws;
    float*    eam    = (float*)w;                                  // 3 floats
    float*    xl     = (float*)(w + 256);
    float*    xr     = xl + (long)N_NODES * HC;
    float*    outacc = xr + (long)N_NODES * HC;
    float*    hbuf   = outacc + (long)N_NODES * HC;
    float*    elog   = hbuf + (long)N_NODES * HIDDEN;
    unsigned* mEnc   = (unsigned*)(elog + (long)E_TOT * HEADS);
    float*    denom  = (float*)(mEnc + (long)N_NODES * HEADS);
    float*    gemb   = denom + (long)N_NODES * HEADS;
    _Float16* wl_s   = (_Float16*)(gemb + NUM_GRAPHS * LATENT);
    _Float16* wr_s   = wl_s + IN_DIM * HC;
    _Float16* wmu_s  = wr_s + IN_DIM * HC;
    _Float16* wdec_s = wmu_s + HIDDEN * LATENT;

    ea_mean_k<<<1, 256, 0, stream>>>(eattr, eam, N_EDGES);

    auto run_layer = [&](const float* xin, int K, const float* Wl, const float* Wr,
                         const float* We, const float* att, const float* bias) {
        int nh = K * HC;
        swizzleB<<<(nh + 255) / 256, 256, 0, stream>>>(Wl, wl_s, K, HC);
        swizzleB<<<(nh + 255) / 256, 256, 0, stream>>>(Wr, wr_s, K, HC);
        // NB=4: 16 N-tiles -> 4 strips per M-tile
        int groups = ((N_NODES + 15) / 16) * ((HC / 16) / 4);
        gemm_wmma<4><<<(groups + 7) / 8, 256, 0, stream>>>(xin, wl_s, nullptr, xl, N_NODES, K, HC);
        gemm_wmma<4><<<(groups + 7) / 8, 256, 0, stream>>>(xin, wr_s, nullptr, xr, N_NODES, K, HC);

        long nm = (long)N_NODES * HEADS;
        fill_u32<<<(int)((nm + 255) / 256), 256, 0, stream>>>(mEnc, 0u, nm);
        fill_u32<<<(int)((nm + 255) / 256), 256, 0, stream>>>((unsigned*)denom, 0u, nm);
        long no = (long)N_NODES * HC;
        fill_u32<<<(int)((no + 255) / 256), 256, 0, stream>>>((unsigned*)outacc, 0u, no);

        edge_logits<<<(E_TOT + 7) / 8, 256, 0, stream>>>(xl, xr, src, dst, eattr, eam,
                                                         We, att, elog, mEnc);
        edge_softmax<<<(int)(((long)E_TOT * HEADS + 255) / 256), 256, 0, stream>>>(dst, mEnc, elog, denom);
        edge_scatter<<<(E_TOT + 7) / 8, 256, 0, stream>>>(xl, src, dst, elog, denom, outacc);
        finalize_h<<<(int)(((long)N_NODES * HIDDEN + 255) / 256), 256, 0, stream>>>(outacc, bias, hbuf);
    };

    run_layer(x,    IN_DIM, Wl0, Wr0, We0, at0, b0);
    run_layer(hbuf, HIDDEN, Wl1, Wr1, We1, at1, b1);

    // z = h @ W_mu + b_mu   (N=32 -> 2 tiles, NB=2)
    swizzleB<<<(HIDDEN * LATENT + 255) / 256, 256, 0, stream>>>(Wmu, wmu_s, HIDDEN, LATENT);
    {
        int groups = (N_NODES / 16) * ((LATENT / 16) / 2);
        gemm_wmma<2><<<(groups + 7) / 8, 256, 0, stream>>>(hbuf, wmu_s, bmu, z, N_NODES, HIDDEN, LATENT);
    }
    // recon = z @ W_dec + b_dec  (N=128 -> 8 tiles, NB=4)
    swizzleB<<<(LATENT * IN_DIM + 255) / 256, 256, 0, stream>>>(Wde, wdec_s, LATENT, IN_DIM);
    {
        int groups = (N_NODES / 16) * ((IN_DIM / 16) / 4);
        gemm_wmma<4><<<(groups + 7) / 8, 256, 0, stream>>>(z, wdec_s, bde, recon, N_NODES, LATENT, IN_DIM);
    }
    // link prediction on original edges
    link_k<<<(N_EDGES + 255) / 256, 256, 0, stream>>>(z, src, dst, wgt, link);
    // graph pooling + classifier
    fill_u32<<<(NUM_GRAPHS * LATENT + 255) / 256, 256, 0, stream>>>((unsigned*)gemb, 0u, NUM_GRAPHS * LATENT);
    pool_k<<<(int)(((long)N_NODES * LATENT + 255) / 256), 256, 0, stream>>>(z, batch, gemb);
    cls_k<<<1, 640, 0, stream>>>(gemb, Wcl, bcl, logitsO);
}
